// MLAAttention_81819126989544
// MI455X (gfx1250) — matrix operand order repaired and checked by reading, not compiled, not run on hardware
//
#include <hip/hip_runtime.h>

typedef __attribute__((ext_vector_type(16))) _Float16 v16h;
typedef __attribute__((ext_vector_type(8)))  _Float16 v8h;
typedef __attribute__((ext_vector_type(4)))  _Float16 v4h;
typedef __attribute__((ext_vector_type(8)))  float    v8f;

#define NH    16
#define NKV   4
#define HD    128
#define RANK  512
#define DIM   2048
#define BATCH 2
#define SEQ   2048
#define EPSQ  1.1920928955078125e-07f

// ---------------------------------------------------------------------------
// WMMA fragment helpers (layouts per CDNA5 ISA §7.12.2, wave32)
// ---------------------------------------------------------------------------
__device__ inline v16h make_v16(v8h lo, v8h hi) {
  v16h r;
#pragma unroll
  for (int i = 0; i < 8; ++i) { r[i] = lo[i]; r[i + 8] = hi[i]; }
  return r;
}

// A-matrix 16x32 f16: lane<16 -> row=lane, K in {0..7,16..23}; lane>=16 -> row=lane-16, K in {8..15,24..31}
__device__ inline v16h load_a_frag(const _Float16* p, int ld) {
  int lane = threadIdx.x & 31;
  const _Float16* rp = p + (lane & 15) * ld + ((lane & 16) ? 8 : 0);
  return make_v16(*(const v8h*)rp, *(const v8h*)(rp + 16));
}

// B-matrix 32x16 f16 (LDS rows = N, cols = K): lane<16 -> col=lane, K in {0..7,16..23}; lane>=16 -> col=lane-16, K in {8..15,24..31}
__device__ inline v16h load_b_frag(const _Float16* p, int ld) {
  int lane = threadIdx.x & 31;
  const _Float16* rp = p + (lane & 15) * ld + ((lane & 16) ? 8 : 0);
  return make_v16(*(const v8h*)rp, *(const v8h*)(rp + 16));
}

__device__ inline v8f wmma_f16(v16h a, v16h b, v8f c) {
  return __builtin_amdgcn_wmma_f32_16x16x32_f16(false, a, false, b, (short)0, c,
                                                false, false);
}

// ---------------------------------------------------------------------------
// Generic GEMM: C[M,N] = A[M,K] * B[N,K]^T   (f32 in/out, f16 WMMA inside)
// Block tile 64(M) x 128(N), K-step 64, double-buffered LDS, 1 barrier/step.
// 256 threads = 8 waves (4 M x 2 N); each wave: 16x64 output, 8 WMMAs/step.
// ---------------------------------------------------------------------------
#define GTM 64
#define GTN 128
#define GTK 64
#define GLD 72  // 64 + pad halves, keeps 16B alignment

__global__ __launch_bounds__(256) void gemm_nt_kernel(
    const float* __restrict__ A, const float* __restrict__ B,
    float* __restrict__ C, int M, int N, int K) {
  __shared__ alignas(16) _Float16 As[2][GTM][GLD];
  __shared__ alignas(16) _Float16 Bs[2][GTN][GLD];
  const int tid = threadIdx.x;
  const int wave = tid >> 5;
  const int lane = tid & 31;
  const int m0 = blockIdx.y * GTM;
  const int n0 = blockIdx.x * GTN;
  const int wm = (wave & 3) * 16;
  const int wn = (wave >> 2) * 64;

  v8f acc[4] = {};
  float4 ra[4], rb[8];  // register staging: all global loads batched

  auto load_tile = [&](int kt) {
#pragma unroll
    for (int j = 0; j < 4; ++j) {  // A: 64 rows x 16 float4
      int idx = tid + j * 256;
      int r = idx >> 4, c4 = idx & 15;
      ra[j] = *(const float4*)(A + (size_t)(m0 + r) * K + kt + c4 * 4);
    }
#pragma unroll
    for (int j = 0; j < 8; ++j) {  // B: 128 rows x 16 float4
      int idx = tid + j * 256;
      int r = idx >> 4, c4 = idx & 15;
      rb[j] = *(const float4*)(B + (size_t)(n0 + r) * K + kt + c4 * 4);
    }
    if (kt + GTK < K) {  // global_prefetch_b8 hint one stage further out
      __builtin_prefetch(A + (size_t)(m0 + (tid & 63)) * K + kt + GTK, 0, 3);
      __builtin_prefetch(B + (size_t)(n0 + (tid & 127)) * K + kt + GTK, 0, 3);
    }
  };
  auto store_tile = [&](int buf) {
#pragma unroll
    for (int j = 0; j < 4; ++j) {
      int idx = tid + j * 256;
      int r = idx >> 4, c4 = idx & 15;
      v4h h = {(_Float16)ra[j].x, (_Float16)ra[j].y, (_Float16)ra[j].z,
               (_Float16)ra[j].w};
      *(v4h*)&As[buf][r][c4 * 4] = h;
    }
#pragma unroll
    for (int j = 0; j < 8; ++j) {
      int idx = tid + j * 256;
      int r = idx >> 4, c4 = idx & 15;
      v4h h = {(_Float16)rb[j].x, (_Float16)rb[j].y, (_Float16)rb[j].z,
               (_Float16)rb[j].w};
      *(v4h*)&Bs[buf][r][c4 * 4] = h;
    }
  };

  load_tile(0);
  store_tile(0);
  const int nT = K / GTK;
  for (int t = 0; t < nT; ++t) {
    __syncthreads();  // buf (t&1) ready; buf ((t+1)&1) fully consumed
    const int cur = t & 1;
    const bool more = (t + 1) < nT;
    if (more) load_tile((t + 1) * GTK);  // global loads drain behind WMMAs
#pragma unroll
    for (int kk = 0; kk < 2; ++kk) {
      // Issue ALL fragment loads with distinct registers, then the WMMA
      // chain: ds loads return in order so wait counts become partial and
      // LDS latency overlaps matrix-pipe issue.
      v16h a  = load_a_frag(&As[cur][wm][kk * 32], GLD);
      v16h b0 = load_b_frag(&Bs[cur][wn +  0][kk * 32], GLD);
      v16h b1 = load_b_frag(&Bs[cur][wn + 16][kk * 32], GLD);
      v16h b2 = load_b_frag(&Bs[cur][wn + 32][kk * 32], GLD);
      v16h b3 = load_b_frag(&Bs[cur][wn + 48][kk * 32], GLD);
      acc[0] = wmma_f16(a, b0, acc[0]);
      acc[1] = wmma_f16(a, b1, acc[1]);
      acc[2] = wmma_f16(a, b2, acc[2]);
      acc[3] = wmma_f16(a, b3, acc[3]);
    }
    if (more) store_tile((t + 1) & 1);
  }

  const int col = lane & 15;
  const int rofs = (lane & 16) ? 8 : 0;
#pragma unroll
  for (int nt = 0; nt < 4; ++nt)
#pragma unroll
    for (int e = 0; e < 8; ++e)
      C[(size_t)(m0 + wm + rofs + e) * N + (n0 + wn + nt * 16 + col)] =
          acc[nt][e];
}

// ---------------------------------------------------------------------------
// RMS-norm + RoPE (+gain) : src f32 [B,T,H,HD] -> dst f16 [B,H,T,HD]
// ---------------------------------------------------------------------------
__global__ __launch_bounds__(256) void prep_qk_kernel(
    const float* __restrict__ src, _Float16* __restrict__ dst,
    const float* __restrict__ gain, int H) {
  const int bt = blockIdx.x;
  const int t = bt & (SEQ - 1);
  const int b = bt >> 11;
  const int wave = threadIdx.x >> 5, lane = threadIdx.x & 31;
  const float kfreq = 0.14391156831212787f;  // log(10000)/64

  for (int h = wave; h < H; h += 8) {
    const float* p = src + ((size_t)bt * H + h) * HD;
    float x0 = p[lane], x1 = p[lane + 32], x2 = p[lane + 64], x3 = p[lane + 96];
    float ss = x0 * x0 + x1 * x1 + x2 * x2 + x3 * x3;
#pragma unroll
    for (int m = 16; m >= 1; m >>= 1) ss += __shfl_xor(ss, m, 32);
    float inv = rsqrtf(ss * (1.0f / 128.0f) + EPSQ);
    float g = gain ? gain[h] : 1.0f;
    x0 *= inv; x1 *= inv; x2 *= inv; x3 *= inv;

    float f0 = __expf(-(float)lane * kfreq);
    float f1 = __expf(-(float)(lane + 32) * kfreq);
    float s0, c0, s1, c1;
    __sincosf((float)t * f0, &s0, &c0);
    __sincosf((float)t * f1, &s1, &c1);

    _Float16* op = dst + (((size_t)b * H + h) * SEQ + t) * HD;
    op[lane]      = (_Float16)((x0 * c0 + x2 * s0) * g);
    op[lane + 32] = (_Float16)((x1 * c1 + x3 * s1) * g);
    op[lane + 64] = (_Float16)((-x0 * s0 + x2 * c0) * g);
    op[lane + 96] = (_Float16)((-x1 * s1 + x3 * c1) * g);
  }
}

// V: f32 [B,T,NKV*HD] -> f16 [B,NKV,T,HD]
__global__ __launch_bounds__(256) void prep_v_kernel(
    const float* __restrict__ src, _Float16* __restrict__ dst) {
  const int bt = blockIdx.x;
  const int t = bt & (SEQ - 1);
  const int b = bt >> 11;
#pragma unroll
  for (int j = 0; j < 2; ++j) {
    int idx = threadIdx.x + j * 256;
    int kv = idx >> 7, d = idx & 127;
    float v = src[(size_t)bt * (NKV * HD) + idx];
    dst[(((size_t)b * NKV + kv) * SEQ + t) * HD + d] = (_Float16)v;
  }
}

// ---------------------------------------------------------------------------
// Flash attention. Block = (b, h, 128 q-rows); 8 waves x 16 q-rows.
// 32-key chunks, double-buffered K/V LDS, 1 barrier/chunk, online softmax.
// ---------------------------------------------------------------------------
#define KSLD 136  // 128 + pad
#define VTLD 40   // 32 + pad
#define PSLD 40

__global__ __launch_bounds__(256) void flash_kernel(
    const _Float16* __restrict__ Q, const _Float16* __restrict__ Km,
    const _Float16* __restrict__ Vm, float* __restrict__ Y) {
  __shared__ alignas(16) _Float16 Ks[2][32][KSLD];   // [key][d]
  __shared__ alignas(16) _Float16 Vt[2][HD][VTLD];   // [d][key] transposed
  __shared__ alignas(16) _Float16 Ps[8][16][PSLD];

  const int b = blockIdx.z, h = blockIdx.y;
  const int kv = h >> 2;  // rep = NH/NKV = 4
  const int q0 = blockIdx.x * 128;
  const int wave = threadIdx.x >> 5, lane = threadIdx.x & 31;
  const int qr0 = q0 + wave * 16;

  const _Float16* qbase = Q + ((size_t)b * NH + h) * SEQ * HD;
  const _Float16* kbase = Km + ((size_t)b * NKV + kv) * SEQ * HD;
  const _Float16* vbase = Vm + ((size_t)b * NKV + kv) * SEQ * HD;

  // Q fragments straight from global (rows contiguous)
  v16h qf[4];
  {
    const _Float16* rp = qbase + (size_t)(qr0 + (lane & 15)) * HD;
    int ofs = (lane & 16) ? 8 : 0;
#pragma unroll
    for (int kk = 0; kk < 4; ++kk)
      qf[kk] = make_v16(*(const v8h*)(rp + kk * 32 + ofs),
                        *(const v8h*)(rp + kk * 32 + ofs + 16));
  }

  float mrow[8], lrow[8];
  v8f o[8] = {};
#pragma unroll
  for (int e = 0; e < 8; ++e) { mrow[e] = -1e30f; lrow[e] = 0.0f; }

  v8h rk[2], rv[2];  // register staging for the next chunk
  auto load_chunk = [&](int kc) {
#pragma unroll
    for (int j = 0; j < 2; ++j) {
      int idx = threadIdx.x + j * 256;  // 512 x v8h over 32x128 halves
      int kr = idx >> 4, d8 = (idx & 15) * 8;
      rk[j] = *(const v8h*)(kbase + (size_t)(kc + kr) * HD + d8);
      rv[j] = *(const v8h*)(vbase + (size_t)(kc + kr) * HD + d8);
    }
  };
  auto store_chunk = [&](int buf) {
#pragma unroll
    for (int j = 0; j < 2; ++j) {
      int idx = threadIdx.x + j * 256;
      int kr = idx >> 4, d8 = (idx & 15) * 8;
      *(v8h*)&Ks[buf][kr][d8] = rk[j];
#pragma unroll
      for (int i = 0; i < 8; ++i) Vt[buf][d8 + i][kr] = rv[j][i];  // transpose
    }
  };

  load_chunk(0);
  store_chunk(0);
  const int nC = (q0 + 128) / 32;  // causal: keys <= last q row of block
  for (int t = 0; t < nC; ++t) {
    const int kc = t * 32;
    __syncthreads();
    const int cur = t & 1;
    const bool more = (t + 1) < nC;
    if (more) load_chunk(kc + 32);

    // S = Q * K^T  (two 16-key column tiles); 2-deep rotating B-frag buffer
    v8f s0 = {}, s1 = {};
    {
      v16h bk0[2], bk1[2];
      bk0[0] = load_b_frag(&Ks[cur][0][0], KSLD);
      bk1[0] = load_b_frag(&Ks[cur][16][0], KSLD);
#pragma unroll
      for (int kk = 0; kk < 4; ++kk) {
        const int c = kk & 1, n = c ^ 1;
        if (kk < 3) {
          bk0[n] = load_b_frag(&Ks[cur][0][(kk + 1) * 32], KSLD);
          bk1[n] = load_b_frag(&Ks[cur][16][(kk + 1) * 32], KSLD);
        }
        s0 = wmma_f16(qf[kk], bk0[c], s0);
        s1 = wmma_f16(qf[kk], bk1[c], s1);
      }
    }

    // scale + causal mask + online softmax
    const float sc = 0.08838834764831845f;  // 1/sqrt(128)
    const int col0 = kc + (lane & 15);
    const int row0 = qr0 + ((lane & 16) ? 8 : 0);
    float pme[8];
#pragma unroll
    for (int e = 0; e < 8; ++e) {
      float a = s0[e] * sc, bb = s1[e] * sc;
      int r = row0 + e;
      if (col0 > r) a = -1e30f;
      if (col0 + 16 > r) bb = -1e30f;
      s0[e] = a; s1[e] = bb;
      pme[e] = fmaxf(a, bb);
    }
#pragma unroll
    for (int m = 8; m >= 1; m >>= 1)
#pragma unroll
      for (int e = 0; e < 8; ++e)
        pme[e] = fmaxf(pme[e], __shfl_xor(pme[e], m, 32));

    float pscale[8], rsum[8];
#pragma unroll
    for (int e = 0; e < 8; ++e) {
      float mn = fmaxf(mrow[e], pme[e]);
      pscale[e] = __expf(mrow[e] - mn);
      mrow[e] = mn;
      float p0 = __expf(s0[e] - mn);
      float p1 = __expf(s1[e] - mn);
      s0[e] = p0; s1[e] = p1;
      rsum[e] = p0 + p1;
    }
#pragma unroll
    for (int m = 8; m >= 1; m >>= 1)
#pragma unroll
      for (int e = 0; e < 8; ++e) rsum[e] += __shfl_xor(rsum[e], m, 32);
#pragma unroll
    for (int e = 0; e < 8; ++e) lrow[e] = lrow[e] * pscale[e] + rsum[e];
#pragma unroll
    for (int dt = 0; dt < 8; ++dt)
#pragma unroll
      for (int e = 0; e < 8; ++e) o[dt][e] *= pscale[e];

    // P (C-layout) -> per-wave LDS -> A-fragment layout (wave-private, in-order)
    {
      int prow = (lane & 16) ? 8 : 0;
      int pcol = lane & 15;
#pragma unroll
      for (int e = 0; e < 8; ++e) {
        Ps[wave][prow + e][pcol] = (_Float16)s0[e];
        Ps[wave][prow + e][pcol + 16] = (_Float16)s1[e];
      }
    }
    {
      v16h ap = load_a_frag(&Ps[wave][0][0], PSLD);
      v16h bv[2];
      bv[0] = load_b_frag(&Vt[cur][0][0], VTLD);
#pragma unroll
      for (int dt = 0; dt < 8; ++dt) {
        const int c = dt & 1, n = c ^ 1;
        if (dt < 7) bv[n] = load_b_frag(&Vt[cur][(dt + 1) * 16][0], VTLD);
        o[dt] = wmma_f16(ap, bv[c], o[dt]);
      }
    }

    if (more) store_chunk((t + 1) & 1);
  }

  // normalize and store y [B,T,NH*HD] f32
  float* ybase = Y + (size_t)b * SEQ * DIM;
  const int col = lane & 15;
  const int rofs = (lane & 16) ? 8 : 0;
#pragma unroll
  for (int e = 0; e < 8; ++e) {
    float invl = 1.0f / lrow[e];
    int trow = qr0 + rofs + e;
#pragma unroll
    for (int dt = 0; dt < 8; ++dt)
      ybase[(size_t)trow * DIM + h * HD + dt * 16 + col] = o[dt][e] * invl;
  }
}

// ---------------------------------------------------------------------------
extern "C" void kernel_launch(void* const* d_in, const int* in_sizes, int n_in,
                              void* d_out, int out_size, void* d_ws,
                              size_t ws_size, hipStream_t stream) {
  (void)in_sizes; (void)n_in; (void)out_size; (void)ws_size;
  const float* x     = (const float*)d_in[0];
  const float* Wq    = (const float*)d_in[1];
  const float* Wdown = (const float*)d_in[2];
  const float* Wkup  = (const float*)d_in[3];
  const float* Wvup  = (const float*)d_in[4];
  const float* Wproj = (const float*)d_in[5];
  const float* qgain = (const float*)d_in[6];
  float* out = (float*)d_out;

  const int M = BATCH * SEQ;  // 4096
  char* ws = (char*)d_ws;
  size_t off = 0;
  auto alloc = [&](size_t bytes) {
    char* p = ws + off;
    off += (bytes + 255) & ~(size_t)255;
    return p;
  };
  float*    lat  = (float*)alloc((size_t)M * RANK * 4);
  float*    qraw = (float*)alloc((size_t)M * DIM * 4);  // reused as y
  float*    kraw = (float*)alloc((size_t)M * NKV * HD * 4);
  float*    vraw = (float*)alloc((size_t)M * NKV * HD * 4);
  _Float16* q16p = (_Float16*)alloc((size_t)M * DIM * 2);
  _Float16* k16p = (_Float16*)alloc((size_t)M * NKV * HD * 2);
  _Float16* v16p = (_Float16*)alloc((size_t)M * NKV * HD * 2);

  dim3 blk(256);
  gemm_nt_kernel<<<dim3(DIM / GTN, M / GTM), blk, 0, stream>>>(x, Wq, qraw, M, DIM, DIM);
  gemm_nt_kernel<<<dim3(RANK / GTN, M / GTM), blk, 0, stream>>>(x, Wdown, lat, M, RANK, DIM);
  gemm_nt_kernel<<<dim3((NKV * HD) / GTN, M / GTM), blk, 0, stream>>>(lat, Wkup, kraw, M, NKV * HD, RANK);
  gemm_nt_kernel<<<dim3((NKV * HD) / GTN, M / GTM), blk, 0, stream>>>(lat, Wvup, vraw, M, NKV * HD, RANK);

  prep_qk_kernel<<<dim3(M), blk, 0, stream>>>(qraw, q16p, qgain, NH);
  prep_qk_kernel<<<dim3(M), blk, 0, stream>>>(kraw, k16p, nullptr, NKV);
  prep_v_kernel<<<dim3(M), blk, 0, stream>>>(vraw, v16p);

  float* y = qraw;  // qraw fully consumed by prep_qk above
  flash_kernel<<<dim3(SEQ / 128, NH, BATCH), blk, 0, stream>>>(q16p, k16p, v16p, y);

  gemm_nt_kernel<<<dim3(DIM / GTN, M / GTM), blk, 0, stream>>>(y, Wproj, out, M, DIM, DIM);
}
